// Encoder1d_16355235463757
// MI455X (gfx1250) — matrix operand order repaired and checked by reading, not compiled, not run on hardware
//
#include <hip/hip_runtime.h>
#include <stdint.h>

typedef __attribute__((ext_vector_type(16))) __bf16 bf16x16;
typedef __attribute__((ext_vector_type(8)))  __bf16 bf16x8;
typedef __attribute__((ext_vector_type(8)))  float  f32x8;

#define N_    128
#define C_    3
#define HW_   128
#define L_    4096
#define D1_   1204
#define D2_   4816
#define M_    (N_ * C_)   /* 384 */
#define D1P   1280        /* padded h1 width / W1p rows (10 * 128)  */
#define K2P   1216        /* padded K for GEMM2 (38 * 32)           */
#define D2P   4864        /* padded h2 width / W2p rows (38 * 128)  */
#define MASKR 0.3f
#define SLOPE_ 0.01f
#define EPS_  1e-5f

// ---------- helpers: fp32 -> bf16(hi) + bf16(lo residual), RNE ----------
__device__ __forceinline__ unsigned short f32_to_bf16_rne(float v) {
  uint32_t u = __float_as_uint(v);
  u += 0x7FFFu + ((u >> 16) & 1u);
  return (unsigned short)(u >> 16);
}
__device__ __forceinline__ float bf16_bits_to_f32(unsigned short h) {
  return __uint_as_float(((uint32_t)h) << 16);
}
__device__ __forceinline__ void split2(float v, unsigned short* hi, unsigned short* lo) {
  unsigned short h = f32_to_bf16_rne(v);
  float r = v - bf16_bits_to_f32(h);
  *hi = h;
  *lo = f32_to_bf16_rne(r);
}

// ---------- stage 0: gather + mask + split ----------
__global__ void gather_split_kernel(const float* __restrict__ x,
                                    const int* __restrict__ coords,
                                    const float* __restrict__ mask,
                                    unsigned short* __restrict__ sHi,
                                    unsigned short* __restrict__ sLo) {
  int idx = blockIdx.x * blockDim.x + threadIdx.x;   // [0, M_*L_)
  int row = idx / L_;                                 // n*C + c
  int l   = idx - row * L_;
  int n   = row / C_;
  int r  = coords[2 * l];
  int cc = coords[2 * l + 1];
  float v = x[((size_t)row * HW_ + r) * HW_ + cc];
  if (mask[(size_t)n * L_ + l] < MASKR) v = 0.f;
  split2(v, &sHi[idx], &sLo[idx]);
}

// ---------- weight conversion with zero padding ----------
__global__ void convert_pad_split_kernel(const float* __restrict__ src,
                                         int srcRows, int srcCols, int dstCols,
                                         unsigned short* __restrict__ hi,
                                         unsigned short* __restrict__ lo) {
  int idx = blockIdx.x * blockDim.x + threadIdx.x;   // [0, dstRows*dstCols)
  int row = idx / dstCols;
  int col = idx - row * dstCols;
  float v = (row < srcRows && col < srcCols) ? src[(size_t)row * srcCols + col] : 0.f;
  split2(v, &hi[idx], &lo[idx]);
}

// ---------- GEMM: C = leaky(A * B^T + bias); bf16x2 split, f32 accumulate ----------
// A: M x K (row-major, hi/lo), B: Ncols x K (row-major, hi/lo), K % 32 == 0.
// Block: 256 threads = 8 waves (2 x 4); block tile 64(M) x 128(N); wave tile 32 x 32.
__launch_bounds__(256)
__global__ void gemm_bf16x2_kernel(const unsigned short* __restrict__ Ahi,
                                   const unsigned short* __restrict__ Alo,
                                   const unsigned short* __restrict__ Bhi,
                                   const unsigned short* __restrict__ Blo,
                                   const float* __restrict__ bias, int biasN,
                                   float* __restrict__ Cmat, int ldc, int K) {
  const int lane = threadIdx.x & 31;
  const int w    = threadIdx.x >> 5;
  const int wm   = w >> 2;          // 0..1
  const int wn   = w & 3;           // 0..3
  const int half = lane >> 4;       // 0..1
  const int lr   = lane & 15;

  const int rowBase = blockIdx.y * 64 + wm * 32;
  const int colBase = blockIdx.x * 128 + wn * 32;

  f32x8 acc[2][2] = {};

  for (int kk = 0; kk < K; kk += 32) {
    bf16x16 aH[2], aL[2], bH[2], bL[2];
    // A fragment: 16x32 bf16; lane half selects a contiguous 16-wide K chunk (same order as B):
    // elems 0..15 -> K = kk + half*16 .. +15 (matches B)
#pragma unroll
    for (int mi = 0; mi < 2; ++mi) {
      size_t base = (size_t)(rowBase + mi * 16 + lr) * K + kk + half * 16;
      bf16x8 h0 = *(const bf16x8*)(Ahi + base);
      bf16x8 h1 = *(const bf16x8*)(Ahi + base + 8);
      aH[mi] = __builtin_shufflevector(h0, h1, 0,1,2,3,4,5,6,7,8,9,10,11,12,13,14,15);
      bf16x8 l0 = *(const bf16x8*)(Alo + base);
      bf16x8 l1 = *(const bf16x8*)(Alo + base + 8);
      aL[mi] = __builtin_shufflevector(l0, l1, 0,1,2,3,4,5,6,7,8,9,10,11,12,13,14,15);
    }
    // B fragment: 32x16 bf16; column = lr, elems 0..15 -> K = kk + half*16 .. +15
#pragma unroll
    for (int ni = 0; ni < 2; ++ni) {
      size_t base = (size_t)(colBase + ni * 16 + lr) * K + kk + half * 16;
      bH[ni] = *(const bf16x16*)(Bhi + base);
      bL[ni] = *(const bf16x16*)(Blo + base);
    }
#pragma unroll
    for (int mi = 0; mi < 2; ++mi)
#pragma unroll
      for (int ni = 0; ni < 2; ++ni) {
        acc[mi][ni] = __builtin_amdgcn_wmma_f32_16x16x32_bf16(
            false, aH[mi], false, bH[ni], (short)0, acc[mi][ni], false, false);
        acc[mi][ni] = __builtin_amdgcn_wmma_f32_16x16x32_bf16(
            false, aL[mi], false, bH[ni], (short)0, acc[mi][ni], false, false);
        acc[mi][ni] = __builtin_amdgcn_wmma_f32_16x16x32_bf16(
            false, aH[mi], false, bL[ni], (short)0, acc[mi][ni], false, false);
      }
  }

  // C/D layout: VGPR r holds M = r (lanes 0-15) / M = 8+r (lanes 16-31), N = lane%16
#pragma unroll
  for (int ni = 0; ni < 2; ++ni) {
    int col = colBase + ni * 16 + lr;
    float bv = (col < biasN) ? bias[col] : 0.f;
#pragma unroll
    for (int mi = 0; mi < 2; ++mi) {
#pragma unroll
      for (int r = 0; r < 8; ++r) {
        int row = rowBase + mi * 16 + half * 8 + r;
        float v = acc[mi][ni][r] + bv;
        v = (v >= 0.f) ? v : SLOPE_ * v;
        Cmat[(size_t)row * ldc + col] = v;
      }
    }
  }
}

// ---------- batchnorm statistics (per channel c over N and valid cols) ----------
__global__ void zero_stats_kernel(float* stats) {
  if (threadIdx.x < 24) stats[threadIdx.x] = 0.f;
}

__global__ void bn_stats_kernel(const float* __restrict__ H, int ldh, int nValid,
                                float* __restrict__ stats) {
  const int c = blockIdx.x;
  long total = (long)N_ * nValid;
  float s1 = 0.f, s2 = 0.f;
  for (long i = (long)blockIdx.y * blockDim.x + threadIdx.x; i < total;
       i += (long)gridDim.y * blockDim.x) {
    int n   = (int)(i / nValid);
    int col = (int)(i - (long)n * nValid);
    float v = H[(size_t)(n * C_ + c) * ldh + col];
    s1 += v;
    s2 += v * v;
  }
  __shared__ float r1[256], r2[256];
  r1[threadIdx.x] = s1;
  r2[threadIdx.x] = s2;
  __syncthreads();
  for (int s = 128; s > 0; s >>= 1) {
    if ((int)threadIdx.x < s) {
      r1[threadIdx.x] += r1[threadIdx.x + s];
      r2[threadIdx.x] += r2[threadIdx.x + s];
    }
    __syncthreads();
  }
  if (threadIdx.x == 0) {
    atomicAdd(&stats[2 * c], r1[0]);
    atomicAdd(&stats[2 * c + 1], r2[0]);
  }
}

__global__ void bn_finalize_kernel(const float* __restrict__ stats, float invCount,
                                   const float* __restrict__ g, const float* __restrict__ be,
                                   float* __restrict__ bnA, float* __restrict__ bnB) {
  int c = threadIdx.x;
  if (c < C_) {
    float mean = stats[2 * c] * invCount;
    float var  = stats[2 * c + 1] * invCount - mean * mean;
    float a = g[c] * rsqrtf(var + EPS_);
    bnA[c] = a;
    bnB[c] = be[c] - mean * a;
  }
}

// BN-apply on h1, re-split to bf16 hi/lo (K padded to K2P with zeros)
__global__ void bn_apply_split_kernel(const float* __restrict__ H,
                                      const float* __restrict__ bnA,
                                      const float* __restrict__ bnB,
                                      unsigned short* __restrict__ hi,
                                      unsigned short* __restrict__ lo) {
  int idx = blockIdx.x * blockDim.x + threadIdx.x;   // [0, M_*K2P)
  int row = idx / K2P;
  int col = idx - row * K2P;
  int c   = row % C_;
  float v = (col < D1_) ? (H[(size_t)row * D1P + col] * bnA[c] + bnB[c]) : 0.f;
  split2(v, &hi[idx], &lo[idx]);
}

// final BN-apply writing the output
__global__ void bn_out_kernel(const float* __restrict__ H2,
                              const float* __restrict__ bnA,
                              const float* __restrict__ bnB,
                              float* __restrict__ out) {
  int idx = blockIdx.x * blockDim.x + threadIdx.x;   // [0, M_*D2_)
  int row = idx / D2_;
  int col = idx - row * D2_;
  int c   = row % C_;
  out[idx] = H2[(size_t)row * D2P + col] * bnA[c] + bnB[c];
}

// ---------- host launcher ----------
extern "C" void kernel_launch(void* const* d_in, const int* in_sizes, int n_in,
                              void* d_out, int out_size, void* d_ws, size_t ws_size,
                              hipStream_t stream) {
  (void)in_sizes; (void)n_in; (void)out_size; (void)ws_size;

  const float* x      = (const float*)d_in[0];
  const int*   coords = (const int*)d_in[1];
  const float* mask   = (const float*)d_in[2];
  const float* W1     = (const float*)d_in[3];
  const float* b1     = (const float*)d_in[4];
  const float* g1     = (const float*)d_in[5];
  const float* be1    = (const float*)d_in[6];
  const float* W2     = (const float*)d_in[7];
  const float* b2     = (const float*)d_in[8];
  const float* g2     = (const float*)d_in[9];
  const float* be2    = (const float*)d_in[10];
  float* out = (float*)d_out;

  char* ws = (char*)d_ws;
  size_t off = 0;
  auto alloc = [&](size_t bytes) -> char* {
    char* p = ws + off;
    off += (bytes + 255) & ~(size_t)255;
    return p;
  };
  unsigned short* W1hi  = (unsigned short*)alloc((size_t)D1P * L_ * 2);
  unsigned short* W1lo  = (unsigned short*)alloc((size_t)D1P * L_ * 2);
  unsigned short* W2hi  = (unsigned short*)alloc((size_t)D2P * K2P * 2);
  unsigned short* W2lo  = (unsigned short*)alloc((size_t)D2P * K2P * 2);
  unsigned short* sHi   = (unsigned short*)alloc((size_t)M_ * L_ * 2);
  unsigned short* sLo   = (unsigned short*)alloc((size_t)M_ * L_ * 2);
  unsigned short* h1nHi = (unsigned short*)alloc((size_t)M_ * K2P * 2);
  unsigned short* h1nLo = (unsigned short*)alloc((size_t)M_ * K2P * 2);
  float* h1p   = (float*)alloc((size_t)M_ * D1P * 4);
  float* h2p   = (float*)alloc((size_t)M_ * D2P * 4);
  float* stats = (float*)alloc(256);
  float* bn1A = stats + 12;
  float* bn1B = stats + 15;
  float* bn2A = stats + 18;
  float* bn2B = stats + 21;

  zero_stats_kernel<<<1, 32, 0, stream>>>(stats);

  // stage 0: gather + mask + bf16 split            (384*4096 / 256 = 6144)
  gather_split_kernel<<<(M_ * L_) / 256, 256, 0, stream>>>(x, coords, mask, sHi, sLo);

  // weight conversions (padded, zero-filled)
  convert_pad_split_kernel<<<(D1P * L_) / 256, 256, 0, stream>>>(W1, D1_, L_, L_, W1hi, W1lo);
  convert_pad_split_kernel<<<(D2P * K2P) / 256, 256, 0, stream>>>(W2, D2_, D1_, K2P, W2hi, W2lo);

  // GEMM1: (384 x 4096) x (1280 x 4096)^T -> h1p (384 x 1280), +b1, leaky
  gemm_bf16x2_kernel<<<dim3(D1P / 128, M_ / 64), 256, 0, stream>>>(
      sHi, sLo, W1hi, W1lo, b1, D1_, h1p, D1P, L_);

  // BN1 stats + finalize
  bn_stats_kernel<<<dim3(C_, 32), 256, 0, stream>>>(h1p, D1P, D1_, stats);
  bn_finalize_kernel<<<1, 32, 0, stream>>>(stats, 1.f / ((float)N_ * (float)D1_), g1, be1, bn1A, bn1B);

  // BN1 apply + split into GEMM2 A operand         (384*1216 / 256 = 1824)
  bn_apply_split_kernel<<<(M_ * K2P) / 256, 256, 0, stream>>>(h1p, bn1A, bn1B, h1nHi, h1nLo);

  // GEMM2: (384 x 1216) x (4864 x 1216)^T -> h2p (384 x 4864), +b2, leaky
  gemm_bf16x2_kernel<<<dim3(D2P / 128, M_ / 64), 256, 0, stream>>>(
      h1nHi, h1nLo, W2hi, W2lo, b2, D2_, h2p, D2P, K2P);

  // BN2 stats + finalize
  bn_stats_kernel<<<dim3(C_, 32), 256, 0, stream>>>(h2p, D2P, D2_, stats + 6);
  bn_finalize_kernel<<<1, 32, 0, stream>>>(stats + 6, 1.f / ((float)N_ * (float)D2_), g2, be2, bn2A, bn2B);

  // BN2 apply -> output                            (384*4816 / 256 = 7224)
  bn_out_kernel<<<(M_ * D2_) / 256, 256, 0, stream>>>(h2p, bn2A, bn2B, out);
}
